// EncoderDecoder_4492535792556
// MI455X (gfx1250) — matrix operand order repaired and checked by reading, not compiled, not run on hardware
//
#include <hip/hip_runtime.h>

// ---------------------------------------------------------------------------
// Model dims (from reference)
// ---------------------------------------------------------------------------
constexpr int B       = 1024;
constexpr int T       = 32;
constexpr int CC      = 32;
constexpr int AD      = 224;   // HGT(7) * CC(32)
constexpr int HID     = 256;
constexpr int VOCAB   = 29;
constexpr int OUT_LEN = 25;

typedef _Float16 v16h __attribute__((ext_vector_type(16)));
typedef _Float16 v8h  __attribute__((ext_vector_type(8)));
typedef float    v8f  __attribute__((ext_vector_type(8)));

// ---------------------------------------------------------------------------
// WMMA fragment loaders (CDNA5 16-bit layouts, wave32).
//   A 16x32 f16: lane l -> row M=l&15, khalf=l>>4.
//     element e (0..15): k = e + 8*khalf + (e>=8 ? 8 : 0)
//     => two contiguous 8-half (16B) runs at k0+8*kh and k0+16+8*kh.
//   B 32x16 f16: lane l -> col N=l&15, khalf=l>>4.
//     element e: k = e + 8*khalf + (e>=8 ? 8 : 0)  (same as A).
//     For C = X @ W^T, B[k][n] = W[n][k]: weight rows load contiguously.
// ---------------------------------------------------------------------------
__device__ __forceinline__ v16h load_a_frag(const _Float16* __restrict__ row, int kh) {
    const v8h* p0 = reinterpret_cast<const v8h*>(row + 8 * kh);
    const v8h* p1 = reinterpret_cast<const v8h*>(row + 16 + 8 * kh);
    v8h lo = *p0, hi = *p1;
    v16h r;
#pragma unroll
    for (int i = 0; i < 8; ++i) { r[i] = lo[i]; r[i + 8] = hi[i]; }
    return r;
}

__device__ __forceinline__ v16h load_b_frag(const _Float16* __restrict__ row, int kh) {
    const v8h* p0 = reinterpret_cast<const v8h*>(row + 8 * kh);
    const v8h* p1 = reinterpret_cast<const v8h*>(row + 16 + 8 * kh);
    v8h lo = *p0, hi = *p1;
    v16h r;
#pragma unroll
    for (int i = 0; i < 8; ++i) { r[i] = lo[i]; r[i + 8] = hi[i]; }
    return r;
}

__device__ __forceinline__ float sigmoidf_(float x) {
    return 1.0f / (1.0f + __expf(-x));
}

// ---------------------------------------------------------------------------
// Fused LSTM step:  g = x @ W1^T + h @ W2^T + bih + bhh ; gate ; update h,c.
// Block = 128 threads (4 waves). Wave w computes the 16x16 tile of gate-chunk
// w (i/f/g/o) for batch rows [m0,m0+16) and hidden cols [d0,d0+16) with WMMA,
// tiles meet in LDS, then gating is applied elementwise.
// h is double-buffered (h16_in -> h16_out); c updates in place (single owner).
// ---------------------------------------------------------------------------
template <int INW, int HIDW>
__global__ void __launch_bounds__(128)
lstm_step_kernel(const _Float16* __restrict__ xbase, size_t x_stride,   // row b: xbase + b*x_stride
                 const _Float16* __restrict__ h16_in,                   // [B, HIDW]
                 const _Float16* __restrict__ W1,                       // [4*HIDW, INW]
                 const _Float16* __restrict__ W2,                       // [4*HIDW, HIDW]
                 const float* __restrict__ bih,
                 const float* __restrict__ bhh,
                 float* __restrict__ c,                                 // [B, HIDW]
                 float* __restrict__ hout_f32, size_t hout_stride,      // row b: + b*hout_stride
                 _Float16* __restrict__ h16_out)                        // [B, HIDW]
{
    static_assert(INW % 32 == 0 && HIDW % 32 == 0, "K must be multiple of 32");

    const int m0   = blockIdx.x * 16;
    const int d0   = blockIdx.y * 16;
    const int wave = threadIdx.x >> 5;   // gate chunk: 0=i 1=f 2=g 3=o
    const int lane = threadIdx.x & 31;
    const int l15  = lane & 15;
    const int kh   = lane >> 4;

    __shared__ float gsm[4][16][16];

    const _Float16* xrow = xbase + (size_t)(m0 + l15) * x_stride;
    const _Float16* hrow = h16_in + (size_t)(m0 + l15) * HIDW;
    const int dg = wave * HIDW + d0 + l15;
    const _Float16* w1row = W1 + (size_t)dg * INW;
    const _Float16* w2row = W2 + (size_t)dg * HIDW;

    v8f acc = {};
#pragma unroll
    for (int kk = 0; kk < INW; kk += 32) {
        v16h a  = load_a_frag(xrow + kk, kh);
        v16h bb = load_b_frag(w1row + kk, kh);
        acc = __builtin_amdgcn_wmma_f32_16x16x32_f16(false, a, false, bb,
                                                     (short)0, acc, false, false);
    }
#pragma unroll
    for (int kk = 0; kk < HIDW; kk += 32) {
        v16h a  = load_a_frag(hrow + kk, kh);
        v16h bb = load_b_frag(w2row + kk, kh);
        acc = __builtin_amdgcn_wmma_f32_16x16x32_f16(false, a, false, bb,
                                                     (short)0, acc, false, false);
    }

    // C/D layout: lane -> N=l15; VGPR r -> M = r + 8*kh
#pragma unroll
    for (int r = 0; r < 8; ++r)
        gsm[wave][r + 8 * kh][l15] = acc[r];
    __syncthreads();

    // 256 gate elements, 128 threads -> 2 each
    for (int e = threadIdx.x; e < 256; e += 128) {
        const int m = e >> 4, n = e & 15;
        const int b = m0 + m, d = d0 + n;
        float gi = gsm[0][m][n] + bih[0 * HIDW + d] + bhh[0 * HIDW + d];
        float gf = gsm[1][m][n] + bih[1 * HIDW + d] + bhh[1 * HIDW + d];
        float gg = gsm[2][m][n] + bih[2 * HIDW + d] + bhh[2 * HIDW + d];
        float go = gsm[3][m][n] + bih[3 * HIDW + d] + bhh[3 * HIDW + d];
        const size_t cidx = (size_t)b * HIDW + d;
        float cn = sigmoidf_(gf) * c[cidx] + sigmoidf_(gi) * tanhf(gg);
        float hn = sigmoidf_(go) * tanhf(cn);
        c[cidx] = cn;
        hout_f32[(size_t)b * hout_stride + d] = hn;
        h16_out[cidx] = (_Float16)hn;
    }
}

// ---------------------------------------------------------------------------
// Direct 3x3 conv (pad 1) + ReLU + non-overlapping max-pool. Channel counts
// (1..32) are too small for efficient K>=32 WMMA, so this stays on VALU.
// ---------------------------------------------------------------------------
__global__ void conv3x3_relu_pool(const float* __restrict__ x, const float* __restrict__ w,
                                  const float* __restrict__ bias, float* __restrict__ y,
                                  int Cin, int Cout, int H, int W, int PH, int PW)
{
    const int OH = H / PH, OW = W / PW;
    const size_t total = (size_t)B * Cout * OH * OW;
    size_t idx = (size_t)blockIdx.x * blockDim.x + threadIdx.x;
    if (idx >= total) return;
    int ow = (int)(idx % OW); size_t t2 = idx / OW;
    int oh = (int)(t2 % OH);  t2 /= OH;
    int co = (int)(t2 % Cout);
    int b  = (int)(t2 / Cout);

    const float* xb = x + (size_t)b * Cin * H * W;
    const float* wc = w + (size_t)co * Cin * 9;
    float m = -3.402823466e38f;
    for (int py = 0; py < PH; ++py) {
        for (int px = 0; px < PW; ++px) {
            const int ih = oh * PH + py, iw = ow * PW + px;
            float acc = bias[co];
            for (int ci = 0; ci < Cin; ++ci) {
                const float* xc = xb + (size_t)ci * H * W;
                const float* wk = wc + ci * 9;
#pragma unroll
                for (int ky = 0; ky < 3; ++ky) {
                    const int yy = ih + ky - 1;
                    if (yy < 0 || yy >= H) continue;
#pragma unroll
                    for (int kx = 0; kx < 3; ++kx) {
                        const int xx = iw + kx - 1;
                        if (xx < 0 || xx >= W) continue;
                        acc += xc[(size_t)yy * W + xx] * wk[ky * 3 + kx];
                    }
                }
            }
            acc = fmaxf(acc, 0.0f);
            m = fmaxf(m, acc);
        }
    }
    y[idx] = m;
}

// conv4 (16->32, H=32, W=7, pool 1x1) writing straight into the transposed
// encoder layout y5[b, t=oh, ow*CC + co] as f16.
__global__ void conv4_to_y5(const float* __restrict__ x, const float* __restrict__ w,
                            const float* __restrict__ bias, _Float16* __restrict__ y5)
{
    const int H = 32, W = 7, Cin = 16, Cout = 32;
    const size_t total = (size_t)B * Cout * H * W;
    size_t idx = (size_t)blockIdx.x * blockDim.x + threadIdx.x;
    if (idx >= total) return;
    int ow = (int)(idx % W); size_t t2 = idx / W;
    int oh = (int)(t2 % H);  t2 /= H;
    int co = (int)(t2 % Cout);
    int b  = (int)(t2 / Cout);

    const float* xb = x + (size_t)b * Cin * H * W;
    const float* wc = w + (size_t)co * Cin * 9;
    float acc = bias[co];
    for (int ci = 0; ci < Cin; ++ci) {
        const float* xc = xb + (size_t)ci * H * W;
        const float* wk = wc + ci * 9;
#pragma unroll
        for (int ky = 0; ky < 3; ++ky) {
            const int yy = oh + ky - 1;
            if (yy < 0 || yy >= H) continue;
#pragma unroll
            for (int kx = 0; kx < 3; ++kx) {
                const int xx = ow + kx - 1;
                if (xx < 0 || xx >= W) continue;
                acc += xc[(size_t)yy * W + xx] * wk[ky * 3 + kx];
            }
        }
    }
    acc = fmaxf(acc, 0.0f);
    y5[((size_t)b * T + oh) * AD + ow * CC + co] = (_Float16)acc;
}

// ---------------------------------------------------------------------------
// Small helpers
// ---------------------------------------------------------------------------
__global__ void f32_to_f16(const float* __restrict__ in, _Float16* __restrict__ out, size_t n) {
    size_t i = (size_t)blockIdx.x * blockDim.x + threadIdx.x;
    if (i < n) out[i] = (_Float16)in[i];
}

__global__ void init_enc_state(const float* __restrict__ h0, const float* __restrict__ c0,
                               _Float16* __restrict__ h16, float* __restrict__ c) {
    size_t i = (size_t)blockIdx.x * blockDim.x + threadIdx.x;
    if (i >= (size_t)B * AD) return;
    int d = (int)(i % AD);
    h16[i] = (_Float16)h0[d];
    c[i]   = c0[d];
}

__global__ void init_dec_state(float* __restrict__ hdec, float* __restrict__ cdec,
                               _Float16* __restrict__ h16) {
    size_t i = (size_t)blockIdx.x * blockDim.x + threadIdx.x;
    if (i >= (size_t)B * HID) return;
    hdec[i] = 0.0f; cdec[i] = 0.0f; h16[i] = (_Float16)0.0f;
}

// enc_score[b,t] = y8[b,t,:] . w_e   (w_e = att_w[HID:])
__global__ void enc_score_kernel(const float* __restrict__ y8, const float* __restrict__ att_w,
                                 float* __restrict__ esc) {
    const int bt = blockIdx.x;       // B*T blocks
    const int tid = threadIdx.x;     // 256 threads
    __shared__ float red[256];
    red[tid] = (tid < AD) ? y8[(size_t)bt * AD + tid] * att_w[HID + tid] : 0.0f;
    __syncthreads();
    for (int s = 128; s > 0; s >>= 1) {
        if (tid < s) red[tid] += red[tid + s];
        __syncthreads();
    }
    if (tid == 0) esc[bt] = red[0];
}

// Per-step attention: softmax over T of enc_score + h.w_h + b, context vector
__global__ void attn_kernel(const float* __restrict__ hdec, const float* __restrict__ y8,
                            const float* __restrict__ esc, const float* __restrict__ att_w,
                            const float* __restrict__ att_b, _Float16* __restrict__ ctx16)
{
    const int b = blockIdx.x;
    const int tid = threadIdx.x;     // 256 threads (== HID)
    __shared__ float red[256];
    __shared__ float alpha[T];

    red[tid] = hdec[(size_t)b * HID + tid] * att_w[tid];   // w_h part
    __syncthreads();
    for (int s = 128; s > 0; s >>= 1) {
        if (tid < s) red[tid] += red[tid + s];
        __syncthreads();
    }
    const float dot = red[0];
    if (tid < T) alpha[tid] = esc[(size_t)b * T + tid] + dot + att_b[0];
    __syncthreads();
    if (tid == 0) {
        float mx = alpha[0];
        for (int t = 1; t < T; ++t) mx = fmaxf(mx, alpha[t]);
        float sum = 0.0f;
        for (int t = 0; t < T; ++t) { float e = __expf(alpha[t] - mx); alpha[t] = e; sum += e; }
        const float inv = 1.0f / sum;
        for (int t = 0; t < T; ++t) alpha[t] *= inv;
    }
    __syncthreads();
    if (tid < AD) {
        float s = 0.0f;
        const float* yb = y8 + (size_t)b * T * AD + tid;
#pragma unroll
        for (int t = 0; t < T; ++t) s += alpha[t] * yb[(size_t)t * AD];
        ctx16[(size_t)b * AD + tid] = (_Float16)s;
    }
}

// out[b, v, s] = h[b,:] . out_w[v,:] + out_b[v]
__global__ void outproj_kernel(const float* __restrict__ hdec, const float* __restrict__ out_w,
                               const float* __restrict__ out_b, float* __restrict__ out, int s)
{
    int idx = blockIdx.x * blockDim.x + threadIdx.x;
    if (idx >= B * VOCAB) return;
    const int b = idx / VOCAB, v = idx % VOCAB;
    float acc = out_b[v];
    const float* h = hdec + (size_t)b * HID;
    const float* w = out_w + (size_t)v * HID;
#pragma unroll 8
    for (int d = 0; d < HID; ++d) acc += h[d] * w[d];
    out[((size_t)b * VOCAB + v) * OUT_LEN + s] = acc;
}

// ---------------------------------------------------------------------------
// Host orchestration
// ---------------------------------------------------------------------------
extern "C" void kernel_launch(void* const* d_in, const int* in_sizes, int n_in,
                              void* d_out, int out_size, void* d_ws, size_t ws_size,
                              hipStream_t stream)
{
    (void)in_sizes; (void)n_in; (void)out_size; (void)ws_size;

    const float* x        = (const float*)d_in[0];
    const float* c1w      = (const float*)d_in[1];
    const float* c1b      = (const float*)d_in[2];
    const float* c2w      = (const float*)d_in[3];
    const float* c2b      = (const float*)d_in[4];
    const float* c3w      = (const float*)d_in[5];
    const float* c3b      = (const float*)d_in[6];
    const float* c4w      = (const float*)d_in[7];
    const float* c4b      = (const float*)d_in[8];
    const float* h0       = (const float*)d_in[9];
    const float* c0       = (const float*)d_in[10];
    const float* lstm_Wih = (const float*)d_in[11];
    const float* lstm_Whh = (const float*)d_in[12];
    const float* lstm_bih = (const float*)d_in[13];
    const float* lstm_bhh = (const float*)d_in[14];
    const float* att_w    = (const float*)d_in[15];
    const float* att_b    = (const float*)d_in[16];
    const float* dec_Wih  = (const float*)d_in[17];
    const float* dec_Whh  = (const float*)d_in[18];
    const float* dec_bih  = (const float*)d_in[19];
    const float* dec_bhh  = (const float*)d_in[20];
    const float* out_w    = (const float*)d_in[21];
    const float* out_b    = (const float*)d_in[22];
    float* out = (float*)d_out;

    // ---- workspace layout (256B aligned), with phase reuse -----------------
    char* ws = (char*)d_ws;
    size_t off = 0;
    auto take = [&](size_t bytes) { size_t o = off; off += (bytes + 255) & ~(size_t)255; return o; };

    const size_t REG = 88080384;           // 1024*16*96*14*4 == 1024*4*192*28*4
    const size_t regA   = take(REG);       // y1, then y3, then y8
    const size_t regB   = take(REG);       // y2, then y5(f16)
    const size_t oWih16 = take((size_t)4 * AD * AD * 2);
    const size_t oWhh16 = take((size_t)4 * AD * AD * 2);
    const size_t odWih  = take((size_t)4 * HID * AD * 2);
    const size_t odWhh  = take((size_t)4 * HID * HID * 2);
    const size_t oh16a  = take((size_t)B * AD * 2);
    const size_t oh16b  = take((size_t)B * AD * 2);
    const size_t oc     = take((size_t)B * AD * 4);
    const size_t ohdec  = take((size_t)B * HID * 4);
    const size_t ocdec  = take((size_t)B * HID * 4);
    const size_t ohd16a = take((size_t)B * HID * 2);
    const size_t ohd16b = take((size_t)B * HID * 2);
    const size_t octx   = take((size_t)B * AD * 2);
    const size_t oesc   = take((size_t)B * T * 4);

    float*    y1     = (float*)(ws + regA);
    float*    y2     = (float*)(ws + regB);
    float*    y3     = (float*)(ws + regA);
    _Float16* y5f16  = (_Float16*)(ws + regB);
    float*    y8     = (float*)(ws + regA);
    _Float16* Wih16  = (_Float16*)(ws + oWih16);
    _Float16* Whh16  = (_Float16*)(ws + oWhh16);
    _Float16* dWih16 = (_Float16*)(ws + odWih);
    _Float16* dWhh16 = (_Float16*)(ws + odWhh);
    _Float16* h16[2] = { (_Float16*)(ws + oh16a), (_Float16*)(ws + oh16b) };
    float*    cbuf   = (float*)(ws + oc);
    float*    hdec   = (float*)(ws + ohdec);
    float*    cdec   = (float*)(ws + ocdec);
    _Float16* hd16[2]= { (_Float16*)(ws + ohd16a), (_Float16*)(ws + ohd16b) };
    _Float16* ctx16  = (_Float16*)(ws + octx);
    float*    esc    = (float*)(ws + oesc);

    auto blocks = [](size_t n, int bs) { return (int)((n + bs - 1) / bs); };

    // ---- conv stack --------------------------------------------------------
    conv3x3_relu_pool<<<blocks((size_t)B * 4 * 192 * 28, 256), 256, 0, stream>>>(
        x, c1w, c1b, y1, 1, 4, 384, 28, 2, 1);
    conv3x3_relu_pool<<<blocks((size_t)B * 16 * 96 * 14, 256), 256, 0, stream>>>(
        y1, c2w, c2b, y2, 4, 16, 192, 28, 2, 2);
    conv3x3_relu_pool<<<blocks((size_t)B * 16 * 32 * 7, 256), 256, 0, stream>>>(
        y2, c3w, c3b, y3, 16, 16, 96, 14, 3, 2);
    conv4_to_y5<<<blocks((size_t)B * 32 * 32 * 7, 256), 256, 0, stream>>>(
        y3, c4w, c4b, y5f16);

    // ---- weight conversion to f16 -----------------------------------------
    f32_to_f16<<<blocks((size_t)4 * AD * AD, 256), 256, 0, stream>>>(lstm_Wih, Wih16, (size_t)4 * AD * AD);
    f32_to_f16<<<blocks((size_t)4 * AD * AD, 256), 256, 0, stream>>>(lstm_Whh, Whh16, (size_t)4 * AD * AD);
    f32_to_f16<<<blocks((size_t)4 * HID * AD, 256), 256, 0, stream>>>(dec_Wih, dWih16, (size_t)4 * HID * AD);
    f32_to_f16<<<blocks((size_t)4 * HID * HID, 256), 256, 0, stream>>>(dec_Whh, dWhh16, (size_t)4 * HID * HID);

    // ---- encoder LSTM (32 sequential WMMA steps) ---------------------------
    init_enc_state<<<blocks((size_t)B * AD, 256), 256, 0, stream>>>(h0, c0, h16[0], cbuf);
    for (int t = 0; t < T; ++t) {
        lstm_step_kernel<AD, AD><<<dim3(B / 16, AD / 16), 128, 0, stream>>>(
            y5f16 + (size_t)t * AD, (size_t)T * AD,
            h16[t & 1], Wih16, Whh16, lstm_bih, lstm_bhh, cbuf,
            y8 + (size_t)t * AD, (size_t)T * AD, h16[(t + 1) & 1]);
    }

    // ---- attention decoder (25 sequential steps) ---------------------------
    enc_score_kernel<<<B * T, 256, 0, stream>>>(y8, att_w, esc);
    init_dec_state<<<blocks((size_t)B * HID, 256), 256, 0, stream>>>(hdec, cdec, hd16[0]);
    for (int s = 0; s < OUT_LEN; ++s) {
        attn_kernel<<<B, 256, 0, stream>>>(hdec, y8, esc, att_w, att_b, ctx16);
        lstm_step_kernel<AD, HID><<<dim3(B / 16, HID / 16), 128, 0, stream>>>(
            ctx16, (size_t)AD,
            hd16[s & 1], dWih16, dWhh16, dec_bih, dec_bhh, cdec,
            hdec, (size_t)HID, hd16[(s + 1) & 1]);
        outproj_kernel<<<blocks((size_t)B * VOCAB, 256), 256, 0, stream>>>(hdec, out_w, out_b, out, s);
    }
}
